// EmbeddingMethod_44169443672299
// MI455X (gfx1250) — matrix operand order repaired and checked by reading, not compiled, not run on hardware
//
#include <hip/hip_runtime.h>
#include <hip/hip_bf16.h>

// ---------------- problem constants ----------------
#define BATCH 2048
#define LDIM  2048              // L
#define NBR   16                // NB
#define HDIM  1024              // H = L/2
#define MTOT  (BATCH * NBR)     // 32768 rows of the big GEMMs
#define SLOPE 0.01f

typedef __attribute__((ext_vector_type(16))) __bf16 bf16x16;
typedef __attribute__((ext_vector_type(8)))  __bf16 bf16x8;
typedef __attribute__((ext_vector_type(8)))  float  f32x8;
typedef __attribute__((ext_vector_type(4)))  unsigned int u32x4;
typedef __attribute__((ext_vector_type(8)))  int i32x8;
typedef __attribute__((ext_vector_type(4)))  int i32x4;

__device__ __forceinline__ float leaky(float v) { return v > 0.0f ? v : SLOPE * v; }

// A operand: lanes 0-15 hold K {0..7,16..23}; lanes 16-31 hold K {8..15,24..31}
// -> per lane: 8 bf16 at p, 8 bf16 at p+16
__device__ __forceinline__ bf16x16 load_a16(const __bf16* p) {
    bf16x8 lo = *(const bf16x8*)p;
    bf16x8 hi = *(const bf16x8*)(p + 16);
    bf16x16 r;
    #pragma unroll
    for (int e = 0; e < 8; ++e) { r[e] = lo[e]; r[8 + e] = hi[e]; }
    return r;
}
// B operand: same K order as A -> per lane: 8 bf16 at p, 8 bf16 at p+16
__device__ __forceinline__ bf16x16 load_b16(const __bf16* p) {
    bf16x8 lo = *(const bf16x8*)p;
    bf16x8 hi = *(const bf16x8*)(p + 16);
    bf16x16 r;
    #pragma unroll
    for (int e = 0; e < 8; ++e) { r[e] = lo[e]; r[8 + e] = hi[e]; }
    return r;
}

// ============================================================
// Kernel 1: row quicknorm (ddof=1) + 16 reflect-pad convolutions
// src row staged into LDS via the Tensor Data Mover (1D tile).
// ============================================================
__global__ __launch_bounds__(256) void norm_conv_kernel(
    const float* __restrict__ src, const float* __restrict__ tgt,
    const float* __restrict__ conv_w, const float* __restrict__ conv_b,
    __bf16* __restrict__ zb)
{
    __shared__ float xr[LDIM];
    __shared__ float red[256];
    __shared__ float cw[NBR * NBR];
    __shared__ float cb[NBR];

    const int b = blockIdx.x;
    const int tid = threadIdx.x;

#if __has_builtin(__builtin_amdgcn_tensor_load_to_lds)
    // ---- TDM: 1D tensor tile (2044 f32) global -> LDS ----
    if (tid < 32) {                       // one wave issues; TDM ignores EXEC
        unsigned long long ga = (unsigned long long)(const void*)(src + (size_t)b * (LDIM - 4));
        unsigned int ldsoff = (unsigned int)(unsigned long long)(void*)xr;
        u32x4 g0;
        g0[0] = 1u;                                        // count=1 (valid user D#)
        g0[1] = ldsoff;                                    // lds_addr
        g0[2] = (unsigned int)(ga & 0xffffffffull);        // global_addr[31:0]
        g0[3] = (unsigned int)((ga >> 32) & 0x1ffffffull)  // global_addr[56:32]
              | (2u << 30);                                // type = 2 ("image")
        i32x8 g1;
        const unsigned int td0 = LDIM - 4;                 // tensor_dim0 = 2044
        g1[0] = (int)(2u << 16);                           // data_size = 4 bytes
        g1[1] = (int)((td0 & 0xffffu) << 16);              // tensor_dim0[15:0]
        g1[2] = (int)(((td0 >> 16) & 0xffffu))             // tensor_dim0[31:16]
              | (int)(1u << 16);                           // tensor_dim1 = 1
        g1[3] = (int)(td0 << 16);                          // tile_dim0 = 2044
        g1[4] = (int)1;                                    // tile_dim1 = 1
        g1[5] = (int)td0;                                  // tensor_dim0_stride[31:0]
        g1[6] = 0;
        g1[7] = 0;
        i32x4 gz4; gz4[0] = 0; gz4[1] = 0; gz4[2] = 0; gz4[3] = 0;
        i32x8 gz8;
        #pragma unroll
        for (int e = 0; e < 8; ++e) gz8[e] = 0;
        __builtin_amdgcn_tensor_load_to_lds(g0, g1, gz4, gz4, gz8, 0);
        __builtin_amdgcn_s_wait_tensorcnt(0);
    }
#else
    for (int l = tid; l < LDIM - 4; l += 256) xr[l] = src[(size_t)b * (LDIM - 4) + l];
#endif
    if (tid < 4)          xr[LDIM - 4 + tid] = tgt[b * 4 + tid];
    if (tid < NBR * NBR)  cw[tid] = conv_w[tid];
    if (tid < NBR)        cb[tid] = conv_b[tid];
    __syncthreads();

    float s = 0.f, s2 = 0.f;
    for (int l = tid; l < LDIM; l += 256) { float v = xr[l]; s += v; s2 += v * v; }
    red[tid] = s; __syncthreads();
    for (int o = 128; o > 0; o >>= 1) { if (tid < o) red[tid] += red[tid + o]; __syncthreads(); }
    const float sum = red[0]; __syncthreads();
    red[tid] = s2; __syncthreads();
    for (int o = 128; o > 0; o >>= 1) { if (tid < o) red[tid] += red[tid + o]; __syncthreads(); }
    const float sumsq = red[0]; __syncthreads();

    const float mu  = sum / (float)LDIM;
    const float var = (sumsq - sum * mu) / (float)(LDIM - 1);   // ddof = 1
    const float inv = rsqrtf(var);

    for (int l = tid; l < LDIM; l += 256) xr[l] = (xr[l] - mu) * inv;
    __syncthreads();

    for (int k = 0; k < NBR; ++k) {
        const int left = k >> 1;
        for (int l = tid; l < LDIM; l += 256) {
            float acc = cb[k];
            for (int i = 0; i <= k; ++i) {
                int idx = l + i - left;                     // reflect (no edge repeat)
                idx = idx < 0 ? -idx : (idx >= LDIM ? 2 * LDIM - 2 - idx : idx);
                acc += xr[idx] * cw[k * NBR + i];
            }
            zb[((size_t)(b * NBR + k)) * LDIM + l] = (__bf16)acc;
        }
    }
}

// ============================================================
// Kernel 2: convert W1, W2 (each H*L f32) to bf16 once
// ============================================================
__global__ __launch_bounds__(256) void cvt_kernel(
    const float* __restrict__ W1, const float* __restrict__ W2,
    __bf16* __restrict__ w1b, __bf16* __restrict__ w2b, int n)
{
    const int i = blockIdx.x * 256 + threadIdx.x;
    if (i < n) { w1b[i] = (__bf16)W1[i]; w2b[i] = (__bf16)W2[i]; }
}

// ============================================================
// Kernel 3: fused MLP. Each block handles 32 rows of M=32768.
// Each wave owns BOTH 16-row halves of the A panel for its
// 128-wide column strip: 8 B-tile loads feed 16 WMMAs per K-step.
// Phase 1: h = leaky(z @ W1^T + b1)  ->  LDS (32x1024 bf16, 64 KB)
// Phase 2: out = z + leaky(h @ W2^T + b2)   (h never touches HBM)
// ============================================================
__global__ __launch_bounds__(256) void mlp_kernel(
    const __bf16* __restrict__ zb,
    const __bf16* __restrict__ w1b, const __bf16* __restrict__ w2b,
    const float* __restrict__ b1, const float* __restrict__ b2,
    float* __restrict__ out)
{
    __shared__ __bf16 hsm[32 * HDIM];                // 64 KB

    const int tid  = threadIdx.x;
    const int wv   = tid >> 5;                       // 0..7
    const int lane = tid & 31;
    const int l16  = lane & 15;
    const int hi   = lane >> 4;                      // 0: lanes 0-15, 1: lanes 16-31
    const int m0   = blockIdx.x * 32;

    // ---------------- phase 1: GEMM1 into LDS ----------------
    {
        const int n0 = wv * 128;                     // 8 waves x 128 = 1024 cols
        f32x8 acc0[8], acc1[8];
        #pragma unroll
        for (int j = 0; j < 8; ++j)
            #pragma unroll
            for (int e = 0; e < 8; ++e) { acc0[j][e] = 0.0f; acc1[j][e] = 0.0f; }

        const __bf16* pa0 = zb + (size_t)(m0 + l16) * LDIM + hi * 8;
        const __bf16* pa1 = zb + (size_t)(m0 + 16 + l16) * LDIM + hi * 8;

        #pragma unroll 1
        for (int k0 = 0; k0 < LDIM; k0 += 32) {
            bf16x16 A0 = load_a16(pa0 + k0);
            bf16x16 A1 = load_a16(pa1 + k0);
            #pragma unroll
            for (int j = 0; j < 8; ++j) {
                const int ncol = n0 + j * 16 + l16;
                bf16x16 Bm = load_b16(w1b + (size_t)ncol * LDIM + k0 + hi * 8);
                acc0[j] = __builtin_amdgcn_wmma_f32_16x16x32_bf16(
                    false, A0, false, Bm, (short)0, acc0[j], false, false);
                acc1[j] = __builtin_amdgcn_wmma_f32_16x16x32_bf16(
                    false, A1, false, Bm, (short)0, acc1[j], false, false);
            }
        }
        // epilogue: bias + leaky -> LDS (D layout: VGPR i = row i / 8+i per half-wave)
        #pragma unroll
        for (int j = 0; j < 8; ++j) {
            const int ncol = n0 + j * 16 + l16;
            const float bias = b1[ncol];
            const int mloc = hi * 8;
            #pragma unroll
            for (int i = 0; i < 8; ++i) {
                hsm[(mloc + i) * HDIM + ncol]      = (__bf16)leaky(acc0[j][i] + bias);
                hsm[(16 + mloc + i) * HDIM + ncol] = (__bf16)leaky(acc1[j][i] + bias);
            }
        }
    }
    __syncthreads();

    // ---------------- phase 2: GEMM2 + residual ----------------
    #pragma unroll 1
    for (int s = 0; s < 2; ++s) {                    // 8 waves x 2 strips x 128 = 2048 cols
        const int n0 = wv * 256 + s * 128;
        f32x8 acc0[8], acc1[8];
        #pragma unroll
        for (int j = 0; j < 8; ++j)
            #pragma unroll
            for (int e = 0; e < 8; ++e) { acc0[j][e] = 0.0f; acc1[j][e] = 0.0f; }

        #pragma unroll 1
        for (int k0 = 0; k0 < HDIM; k0 += 32) {
            bf16x16 A0 = load_a16(hsm + (l16) * HDIM + k0 + hi * 8);       // ds path
            bf16x16 A1 = load_a16(hsm + (16 + l16) * HDIM + k0 + hi * 8);  // ds path
            #pragma unroll
            for (int j = 0; j < 8; ++j) {
                const int ncol = n0 + j * 16 + l16;            // out column l
                bf16x16 Bm = load_b16(w2b + (size_t)ncol * HDIM + k0 + hi * 8);
                acc0[j] = __builtin_amdgcn_wmma_f32_16x16x32_bf16(
                    false, A0, false, Bm, (short)0, acc0[j], false, false);
                acc1[j] = __builtin_amdgcn_wmma_f32_16x16x32_bf16(
                    false, A1, false, Bm, (short)0, acc1[j], false, false);
            }
        }
        #pragma unroll
        for (int j = 0; j < 8; ++j) {
            const int ncol = n0 + j * 16 + l16;
            const float bias = b2[ncol];
            const int mb0 = m0 + hi * 8;
            #pragma unroll
            for (int i = 0; i < 8; ++i) {
                const size_t idx0 = (size_t)(mb0 + i) * LDIM + ncol;
                const size_t idx1 = (size_t)(mb0 + 16 + i) * LDIM + ncol;
                out[idx0] = (float)zb[idx0] + leaky(acc0[j][i] + bias);
                out[idx1] = (float)zb[idx1] + leaky(acc1[j][i] + bias);
            }
        }
    }
}

// ============================================================
// launcher
// ============================================================
extern "C" void kernel_launch(void* const* d_in, const int* in_sizes, int n_in,
                              void* d_out, int out_size, void* d_ws, size_t ws_size,
                              hipStream_t stream)
{
    const float* src    = (const float*)d_in[0];
    const float* tgt    = (const float*)d_in[1];
    const float* conv_w = (const float*)d_in[2];
    const float* conv_b = (const float*)d_in[3];
    const float* W1     = (const float*)d_in[4];
    const float* b1     = (const float*)d_in[5];
    const float* W2     = (const float*)d_in[6];
    const float* b2     = (const float*)d_in[7];
    float* out = (float*)d_out;

    // workspace layout: z bf16 (134 MB) | W1 bf16 (4 MB) | W2 bf16 (4 MB)
    __bf16* zb  = (__bf16*)d_ws;
    __bf16* w1b = (__bf16*)((char*)d_ws + (size_t)MTOT * LDIM * sizeof(__bf16));
    __bf16* w2b = w1b + (size_t)HDIM * LDIM;

    norm_conv_kernel<<<BATCH, 256, 0, stream>>>(src, tgt, conv_w, conv_b, zb);

    const int nW = HDIM * LDIM;
    cvt_kernel<<<(nW + 255) / 256, 256, 0, stream>>>(W1, W2, w1b, w2b, nW);

    mlp_kernel<<<MTOT / 32, 256, 0, stream>>>(zb, w1b, w2b, b1, b2, out);
}
